// BidirectMixerModel_3307124818496
// MI455X (gfx1250) — matrix operand order repaired and checked by reading, not compiled, not run on hardware
//
#include <hip/hip_runtime.h>
#include <hip/hip_bf16.h>

// ---------------- constants ----------------
#define D_MODEL 768
#define D_INNER 1536
#define D_STATE 16
#define D_CONV  4
#define DT_RANK 48
#define DT_PAD  64
#define N_LAYERS 4
#define BATCH 4
#define SEQ 1024
#define MROWS (BATCH * SEQ)          // 4096
#define NXD (DT_RANK + 2 * D_STATE)  // 80

typedef __attribute__((ext_vector_type(16))) _Float16 v16h;
typedef __attribute__((ext_vector_type(8)))  float    v8f;
typedef __attribute__((ext_vector_type(4)))  unsigned v4u;
typedef __attribute__((ext_vector_type(4)))  unsigned uint32x4;
typedef __attribute__((ext_vector_type(8)))  int      int32x8;
typedef __attribute__((ext_vector_type(4)))  int      int32x4;

union Frag16 { v16h h; v4u q[2]; unsigned u[8]; };

__device__ __forceinline__ float softplus_fast(float v) {
    // hardware trans ops only (v_exp_f32 / v_log_f32); clamp avoids overflow
    return (v > 20.f) ? v : __logf(1.f + __expf(v));
}

// ---------------- elementwise converts ----------------
__global__ void cvt_f16_kernel(const float* __restrict__ src, _Float16* __restrict__ dst, int n) {
    int i = blockIdx.x * 256 + threadIdx.x;
    if (i < n) dst[i] = (_Float16)src[i];
}

// dt_proj_w (8,1536,48) -> f16 padded (8,1536,64), zeros in cols 48..63
__global__ void cvt_dtw_kernel(const float* __restrict__ src, _Float16* __restrict__ dst, int n) {
    int i = blockIdx.x * 256 + threadIdx.x;
    if (i >= n) return;
    int col = i & 63;
    int row = i >> 6;                      // row over 8*1536
    dst[i] = (col < DT_RANK) ? (_Float16)src[row * DT_RANK + col] : (_Float16)0.f;
}

// xdbl (M,80) cols 0..47 -> f16 padded (M,64)
__global__ void dt_extract_kernel(const float* __restrict__ xdbl, _Float16* __restrict__ dt16, int n) {
    int i = blockIdx.x * 256 + threadIdx.x;
    if (i >= n) return;
    int col = i & 63;
    int row = i >> 6;
    dt16[i] = (col < DT_RANK) ? (_Float16)xdbl[(size_t)row * NXD + col] : (_Float16)0.f;
}

__global__ void add2_kernel(const float* __restrict__ a, const float* __restrict__ b,
                            float* __restrict__ c, int n) {
    int i = blockIdx.x * 256 + threadIdx.x;
    if (i < n) c[i] = a[i] + b[i];
}

// ---------------- residual add (+optional seq flip) + LayerNorm ----------------
__global__ void addln_kernel(const float* __restrict__ hin, const float* __restrict__ resin,
                             int flip, const float* __restrict__ w, const float* __restrict__ bb,
                             float* __restrict__ res_out, _Float16* __restrict__ out16,
                             float* __restrict__ out32) {
    int row  = blockIdx.x;
    int bidx = row >> 10;
    int l    = row & (SEQ - 1);
    int hrow = flip ? ((bidx << 10) + (SEQ - 1 - l)) : row;
    const float* hp = hin + (size_t)hrow * D_MODEL;
    const float* rp = resin ? (resin + (size_t)row * D_MODEL) : nullptr;
    int tid = threadIdx.x;

    __shared__ float red[8];
    __shared__ float stat;

    float v[3];
    float sum = 0.f;
#pragma unroll
    for (int i = 0; i < 3; ++i) {
        int c = tid + i * 256;
        float x = hp[c] + (rp ? rp[c] : 0.f);
        v[i] = x; sum += x;
    }
    for (int o = 16; o > 0; o >>= 1) sum += __shfl_xor(sum, o);
    if ((tid & 31) == 0) red[tid >> 5] = sum;
    __syncthreads();
    if (tid == 0) { float s = 0.f; for (int i = 0; i < 8; ++i) s += red[i]; stat = s; }
    __syncthreads();
    float mu = stat * (1.f / D_MODEL);

    float var = 0.f;
#pragma unroll
    for (int i = 0; i < 3; ++i) { float d = v[i] - mu; var += d * d; }
    for (int o = 16; o > 0; o >>= 1) var += __shfl_xor(var, o);
    if ((tid & 31) == 0) red[tid >> 5] = var;
    __syncthreads();
    if (tid == 0) { float s = 0.f; for (int i = 0; i < 8; ++i) s += red[i]; stat = s; }
    __syncthreads();
    float rstd = rsqrtf(stat * (1.f / D_MODEL) + 1e-5f);

#pragma unroll
    for (int i = 0; i < 3; ++i) {
        int c = tid + i * 256;
        size_t idx = (size_t)row * D_MODEL + c;
        if (res_out) res_out[idx] = v[i];
        float xn = (v[i] - mu) * rstd * w[c] + bb[c];
        if (out16) out16[idx] = (_Float16)xn;
        if (out32) out32[idx] = xn;
    }
}

// ---------------- WMMA GEMM:  C[M,N] = A[M,K](f16,row) x W[N,K](f16,row)^T ----------------
// grid = (ceil(N/512), M/32), block = 256 (8 waves). Each wave computes a 32x64 C
// macro-tile: 2 M-fragments x 4 N-tiles = 8 independent wmma chains per k chunk.
// mode: 0 = store, 1 = accumulate+store, 2 = softplus(val + bias[col]) store.
__global__ __launch_bounds__(256) void wmma_gemm_kernel(
    const _Float16* __restrict__ A, int lda,
    const _Float16* __restrict__ W, int ldw,
    float* __restrict__ C, int ldc,
    int N, int K, int mode, const float* __restrict__ bias) {
    const int tid  = threadIdx.x;
    const int wave = tid >> 5;
    const int lane = tid & 31;
    const int m0   = blockIdx.y << 5;                 // 32 rows per block
    const int n0   = (blockIdx.x << 9) + (wave << 6); // 64 cols per wave
    if (n0 >= N) return;                              // wave fully inactive (no barriers here)

    bool act1 = (n0 + 16 < N);
    bool act2 = (n0 + 32 < N);
    bool act3 = (n0 + 48 < N);
    const bool full = act3;

    const int mrow  = lane & 15;
    const int khalf = lane >> 4;
    const int nn    = n0 + (lane & 15);

    const _Float16* ap0 = A + (size_t)(m0 + mrow) * lda + (khalf << 4);
    const _Float16* ap1 = ap0 + (size_t)16 * lda;
    const _Float16* wp0 = W + (size_t)nn * ldw + (khalf << 4);
    const _Float16* wp1 = wp0 + (size_t)16 * ldw;
    const _Float16* wp2 = wp0 + (size_t)32 * ldw;
    const _Float16* wp3 = wp0 + (size_t)48 * ldw;

    v8f acc[2][4] = {};
    if (full) {
        for (int k0 = 0; k0 < K; k0 += 32) {
            Frag16 a0, a1, b0, b1, b2, b3;
            a0.q[0] = *(const v4u*)(ap0 + k0);
            a0.q[1] = *(const v4u*)(ap0 + k0 + 8);
            a1.q[0] = *(const v4u*)(ap1 + k0);
            a1.q[1] = *(const v4u*)(ap1 + k0 + 8);
            b0.q[0] = *(const v4u*)(wp0 + k0);
            b0.q[1] = *(const v4u*)(wp0 + k0 + 8);
            b1.q[0] = *(const v4u*)(wp1 + k0);
            b1.q[1] = *(const v4u*)(wp1 + k0 + 8);
            b2.q[0] = *(const v4u*)(wp2 + k0);
            b2.q[1] = *(const v4u*)(wp2 + k0 + 8);
            b3.q[0] = *(const v4u*)(wp3 + k0);
            b3.q[1] = *(const v4u*)(wp3 + k0 + 8);
            __builtin_prefetch(ap0 + k0 + 64, 0, 1);   // -> global_prefetch_b8
            __builtin_prefetch(ap1 + k0 + 64, 0, 1);
            __builtin_prefetch(wp0 + k0 + 64, 0, 1);
            __builtin_prefetch(wp1 + k0 + 64, 0, 1);
            __builtin_prefetch(wp2 + k0 + 64, 0, 1);
            __builtin_prefetch(wp3 + k0 + 64, 0, 1);
            acc[0][0] = __builtin_amdgcn_wmma_f32_16x16x32_f16(false, a0.h, false, b0.h, (short)0, acc[0][0], false, false);
            acc[0][1] = __builtin_amdgcn_wmma_f32_16x16x32_f16(false, a0.h, false, b1.h, (short)0, acc[0][1], false, false);
            acc[0][2] = __builtin_amdgcn_wmma_f32_16x16x32_f16(false, a0.h, false, b2.h, (short)0, acc[0][2], false, false);
            acc[0][3] = __builtin_amdgcn_wmma_f32_16x16x32_f16(false, a0.h, false, b3.h, (short)0, acc[0][3], false, false);
            acc[1][0] = __builtin_amdgcn_wmma_f32_16x16x32_f16(false, a1.h, false, b0.h, (short)0, acc[1][0], false, false);
            acc[1][1] = __builtin_amdgcn_wmma_f32_16x16x32_f16(false, a1.h, false, b1.h, (short)0, acc[1][1], false, false);
            acc[1][2] = __builtin_amdgcn_wmma_f32_16x16x32_f16(false, a1.h, false, b2.h, (short)0, acc[1][2], false, false);
            acc[1][3] = __builtin_amdgcn_wmma_f32_16x16x32_f16(false, a1.h, false, b3.h, (short)0, acc[1][3], false, false);
        }
    } else {
        for (int k0 = 0; k0 < K; k0 += 32) {
            Frag16 a0, a1, b;
            a0.q[0] = *(const v4u*)(ap0 + k0);
            a0.q[1] = *(const v4u*)(ap0 + k0 + 8);
            a1.q[0] = *(const v4u*)(ap1 + k0);
            a1.q[1] = *(const v4u*)(ap1 + k0 + 8);
            b.q[0] = *(const v4u*)(wp0 + k0);
            b.q[1] = *(const v4u*)(wp0 + k0 + 8);
            acc[0][0] = __builtin_amdgcn_wmma_f32_16x16x32_f16(false, a0.h, false, b.h, (short)0, acc[0][0], false, false);
            acc[1][0] = __builtin_amdgcn_wmma_f32_16x16x32_f16(false, a1.h, false, b.h, (short)0, acc[1][0], false, false);
            if (act1) {
                b.q[0] = *(const v4u*)(wp1 + k0);
                b.q[1] = *(const v4u*)(wp1 + k0 + 8);
                acc[0][1] = __builtin_amdgcn_wmma_f32_16x16x32_f16(false, a0.h, false, b.h, (short)0, acc[0][1], false, false);
                acc[1][1] = __builtin_amdgcn_wmma_f32_16x16x32_f16(false, a1.h, false, b.h, (short)0, acc[1][1], false, false);
            }
            if (act2) {
                b.q[0] = *(const v4u*)(wp2 + k0);
                b.q[1] = *(const v4u*)(wp2 + k0 + 8);
                acc[0][2] = __builtin_amdgcn_wmma_f32_16x16x32_f16(false, a0.h, false, b.h, (short)0, acc[0][2], false, false);
                acc[1][2] = __builtin_amdgcn_wmma_f32_16x16x32_f16(false, a1.h, false, b.h, (short)0, acc[1][2], false, false);
            }
        }
    }

    // Epilogue: lanes 0-15 -> M = r, lanes 16-31 -> M = 8 + r  (16x16 f32 C layout)
    const int mb = m0 + ((lane >> 4) << 3);
#pragma unroll
    for (int mi = 0; mi < 2; ++mi) {
#pragma unroll
        for (int t = 0; t < 4; ++t) {
            bool act = (t == 0) || (t == 1 && act1) || (t == 2 && act2) || (t == 3 && act3);
            if (!act) continue;
            int ncol = nn + (t << 4);
            float bcol = (mode == 2) ? bias[ncol] : 0.f;
#pragma unroll
            for (int r = 0; r < 8; ++r) {
                size_t idx = (size_t)(mb + (mi << 4) + r) * ldc + ncol;
                float val = acc[mi][t][r];
                if (mode == 2) {
                    C[idx] = softplus_fast(val + bcol);         // fused softplus epilogue
                } else if (mode == 1) {
                    C[idx] = C[idx] + val;
                } else {
                    C[idx] = val;
                }
            }
        }
    }
}

// ---------------- depthwise causal conv (width 4) + bias + SiLU ----------------
__global__ void conv_silu_kernel(const float* __restrict__ xz,
                                 const float* __restrict__ cw, const float* __restrict__ cb,
                                 float* __restrict__ xf, _Float16* __restrict__ xh) {
    int c   = blockIdx.x * 256 + threadIdx.x;
    int l   = blockIdx.y;
    int bdx = blockIdx.z;
    float s = cb[c];
#pragma unroll
    for (int t = 0; t < D_CONV; ++t) {
        int ls = l - (D_CONV - 1) + t;
        if (ls >= 0)
            s += cw[c * D_CONV + t] * xz[(size_t)(bdx * SEQ + ls) * (2 * D_INNER) + c];
    }
    float o = s / (1.f + __expf(-s));
    size_t idx = (size_t)(bdx * SEQ + l) * D_INNER + c;
    xf[idx] = o;
    xh[idx] = (_Float16)o;
}

// ---------------- selective scan, fused D-skip + SiLU(z) gate ----------------
// 16 lanes per channel; grid (96, BATCH), block 256 (16 channels). B/C tiles are
// DMA'd global->LDS by the Tensor Data Mover (double-buffered so the DMA of chunk
// c+1 overlaps the 16 serial scan steps of chunk c); wave 0 issues & waits TENSORcnt.
__global__ __launch_bounds__(256) void scan_kernel(
    const float* __restrict__ delta, const float* __restrict__ xf,
    const float* __restrict__ xdbl, const float* __restrict__ xz,
    const float* __restrict__ A_log, const float* __restrict__ Dp,
    _Float16* __restrict__ y16) {
    __shared__ float BC[2][16][32];           // double-buffered 16 steps x (B[16]|C[16])
    int tid = threadIdx.x;
    int n   = tid & 15;                       // state index
    int d   = blockIdx.x * 16 + (tid >> 4);   // channel
    int bdx = blockIdx.y;

    float An = -__expf(A_log[d * D_STATE + n]);
    float Dv = Dp[d];
    float s  = 0.f;

#if __has_builtin(__builtin_amdgcn_tensor_load_to_lds)
    const bool issuer = (tid < 32);           // wave 0 drives the TDM
    // Loop-invariant descriptor pieces (D# per CDNA5 ISA ch.8):
    // 2D tile, 4B elements, 32x16 tile, row stride 80 elements.
    int32x8 g1;
    g1[0] = 0x00020000;      // data_size=2 (4B), no multicast/pad/iterate
    g1[1] = 32 << 16;        // tensor_dim0 = 32 elements
    g1[2] = 16 << 16;        // tensor_dim1 = 16 rows
    g1[3] = 32 << 16;        // tile_dim0 = 32
    g1[4] = 16;              // tile_dim1 = 16, tile_dim2 = 0
    g1[5] = NXD;             // tensor_dim0_stride = 80 elements
    g1[6] = 0;
    g1[7] = 0;
    const int32x4 gz4 = {0, 0, 0, 0};
    const int32x8 gz8 = {0, 0, 0, 0, 0, 0, 0, 0};
    const unsigned long long gbase = (unsigned long long)
        (xdbl + (size_t)(bdx * SEQ) * NXD + DT_RANK);
    auto issue = [&](int l0, int buf) {
        unsigned lds = (unsigned)(unsigned long long)(&BC[buf][0][0]); // flat low 32b = LDS addr
        unsigned long long ga = gbase + (unsigned long long)l0 * (NXD * 4);
        uint32x4 g0;
        g0[0] = 1u;                                   // count=1 valid descriptor
        g0[1] = lds;                                  // lds_addr
        g0[2] = (unsigned)ga;                         // global_addr[31:0]
        g0[3] = (unsigned)(ga >> 32) | (2u << 30);    // global_addr[56:32] | type=2
        __builtin_amdgcn_tensor_load_to_lds(g0, g1, gz4, gz4, gz8, 0);  // 6-arg form
    };
    if (issuer) issue(0, 0);
#pragma unroll 1
    for (int c = 0; c < SEQ / 16; ++c) {
        int l0  = c << 4;
        int buf = c & 1;
        if (issuer) __builtin_amdgcn_s_wait_tensorcnt(0);   // chunk c resident
        __syncthreads();
        if (issuer && (l0 + 16 < SEQ)) issue(l0 + 16, buf ^ 1); // DMA overlaps compute
#else
#pragma unroll 1
    for (int c = 0; c < SEQ / 16; ++c) {
        int l0  = c << 4;
        int buf = c & 1;
        for (int e = tid; e < 512; e += 256) {
            int rr = e >> 5, cc = e & 31;
            BC[buf][rr][cc] = xdbl[(size_t)(bdx * SEQ + l0 + rr) * NXD + DT_RANK + cc];
        }
        __syncthreads();
#endif
#pragma unroll 4
        for (int i = 0; i < 16; ++i) {
            size_t idxm = (size_t)(bdx * SEQ + l0 + i) * D_INNER + d;
            float dl = delta[idxm];
            float u  = xf[idxm];
            s = __expf(dl * An) * s + (dl * u) * BC[buf][i][n];
            float y = s * BC[buf][i][16 + n];
            y += __shfl_xor(y, 1);
            y += __shfl_xor(y, 2);
            y += __shfl_xor(y, 4);
            y += __shfl_xor(y, 8);
            if (n == 0) {
                y += u * Dv;
                float z = xz[(size_t)(bdx * SEQ + l0 + i) * (2 * D_INNER) + D_INNER + d];
                y *= z / (1.f + __expf(-z));
                y16[idxm] = (_Float16)y;
            }
        }
        __syncthreads();
    }
}

// ---------------- host orchestration ----------------
extern "C" void kernel_launch(void* const* d_in, const int* in_sizes, int n_in,
                              void* d_out, int out_size, void* d_ws, size_t ws_size,
                              hipStream_t stream) {
    const float* input     = (const float*)d_in[0];
    const float* norm_w    = (const float*)d_in[1];
    const float* norm_b    = (const float*)d_in[2];
    const float* in_proj_w = (const float*)d_in[3];
    const float* conv_w    = (const float*)d_in[4];
    const float* conv_b    = (const float*)d_in[5];
    const float* x_proj_w  = (const float*)d_in[6];
    const float* dt_proj_w = (const float*)d_in[7];
    const float* dt_proj_b = (const float*)d_in[8];
    const float* A_log     = (const float*)d_in[9];
    const float* D_skip    = (const float*)d_in[10];
    const float* out_proj_w= (const float*)d_in[11];
    const float* norm_f_w  = (const float*)d_in[12];
    const float* norm_f_b  = (const float*)d_in[13];

    char* ws = (char*)d_ws;
    size_t off = 0;
    auto carve = [&](size_t bytes) -> void* {
        void* p = ws + off;
        off += (bytes + 255) & ~(size_t)255;
        return p;
    };

    const int LD = N_LAYERS * 2;
    _Float16* w_in16  = (_Float16*)carve((size_t)LD * 2 * D_INNER * D_MODEL * 2);
    _Float16* w_x16   = (_Float16*)carve((size_t)LD * NXD * D_INNER * 2);
    _Float16* w_dt16  = (_Float16*)carve((size_t)LD * D_INNER * DT_PAD * 2);
    _Float16* w_out16 = (_Float16*)carve((size_t)LD * D_MODEL * D_INNER * 2);
    float* hA   = (float*)carve((size_t)MROWS * D_MODEL * 4);
    float* hB   = (float*)carve((size_t)MROWS * D_MODEL * 4);
    float* resb = (float*)carve((size_t)MROWS * D_MODEL * 4);
    float* r1   = (float*)carve((size_t)MROWS * D_MODEL * 4);
    float* r2   = (float*)carve((size_t)MROWS * D_MODEL * 4);
    _Float16* hn16 = (_Float16*)carve((size_t)MROWS * D_MODEL * 2);
    float* xz   = (float*)carve((size_t)MROWS * 2 * D_INNER * 4);
    float* xf32 = (float*)carve((size_t)MROWS * D_INNER * 4);
    _Float16* xf16 = (_Float16*)carve((size_t)MROWS * D_INNER * 2);
    float* xdbl = (float*)carve((size_t)MROWS * NXD * 4);
    _Float16* dt16 = (_Float16*)carve((size_t)MROWS * DT_PAD * 2);
    float* dtproj = (float*)carve((size_t)MROWS * D_INNER * 4);
    _Float16* y16 = (_Float16*)carve((size_t)MROWS * D_INNER * 2);
    (void)ws_size; (void)in_sizes; (void)n_in; (void)out_size;

    // ---- weight conversion to f16 (L2-resident thereafter) ----
    {
        int n;
        n = LD * 2 * D_INNER * D_MODEL;
        cvt_f16_kernel<<<(n + 255) / 256, 256, 0, stream>>>(in_proj_w, w_in16, n);
        n = LD * NXD * D_INNER;
        cvt_f16_kernel<<<(n + 255) / 256, 256, 0, stream>>>(x_proj_w, w_x16, n);
        n = LD * D_MODEL * D_INNER;
        cvt_f16_kernel<<<(n + 255) / 256, 256, 0, stream>>>(out_proj_w, w_out16, n);
        n = LD * D_INNER * DT_PAD;
        cvt_dtw_kernel<<<(n + 255) / 256, 256, 0, stream>>>(dt_proj_w, w_dt16, n);
    }

    const float* hcur = input;
    for (int j = 0; j < N_LAYERS; ++j) {
        float* hnew = (j & 1) ? hB : hA;
        for (int d = 0; d < 2; ++d) {
            int ld = j * 2 + d;
            // residual + LN (+flip for reverse dir); emits f16 A-matrix for in_proj
            addln_kernel<<<MROWS, 256, 0, stream>>>(
                hcur, (j == 0) ? nullptr : resb, d,
                norm_w + (size_t)ld * D_MODEL, norm_b + (size_t)ld * D_MODEL,
                (d == 0) ? r1 : r2, hn16, nullptr);
            // in_proj: (4096x768) x (3072x768)^T -> xz
            wmma_gemm_kernel<<<dim3(2 * D_INNER / 512, MROWS / 32), 256, 0, stream>>>(
                hn16, D_MODEL, w_in16 + (size_t)ld * 2 * D_INNER * D_MODEL, D_MODEL,
                xz, 2 * D_INNER, 2 * D_INNER, D_MODEL, 0, nullptr);
            // depthwise conv + bias + SiLU
            conv_silu_kernel<<<dim3(D_INNER / 256, SEQ, BATCH), 256, 0, stream>>>(
                xz, conv_w + (size_t)ld * D_INNER * D_CONV, conv_b + (size_t)ld * D_INNER,
                xf32, xf16);
            // x_proj: (4096x1536) x (80x1536)^T -> xdbl (partial tiles masked)
            wmma_gemm_kernel<<<dim3(1, MROWS / 32), 256, 0, stream>>>(
                xf16, D_INNER, w_x16 + (size_t)ld * NXD * D_INNER, D_INNER,
                xdbl, NXD, NXD, D_INNER, 0, nullptr);
            // dt slice -> padded f16
            dt_extract_kernel<<<(MROWS * DT_PAD) / 256, 256, 0, stream>>>(
                xdbl, dt16, MROWS * DT_PAD);
            // dt_proj: (4096x64) x (1536x64)^T -> delta, fused softplus(+bias) epilogue
            wmma_gemm_kernel<<<dim3(D_INNER / 512, MROWS / 32), 256, 0, stream>>>(
                dt16, DT_PAD, w_dt16 + (size_t)ld * D_INNER * DT_PAD, DT_PAD,
                dtproj, D_INNER, D_INNER, DT_PAD, 2, dt_proj_b + (size_t)ld * D_INNER);
            // selective scan (TDM-staged B/C), fused skip + gate -> y16
            scan_kernel<<<dim3(D_INNER / 16, BATCH), 256, 0, stream>>>(
                dtproj, xf32, xdbl, xz,
                A_log + (size_t)ld * D_INNER * D_STATE, D_skip + (size_t)ld * D_INNER, y16);
            // out_proj: (4096x1536) x (768x1536)^T -> hnew (dir1 accumulates)
            wmma_gemm_kernel<<<dim3((D_MODEL + 511) / 512, MROWS / 32), 256, 0, stream>>>(
                y16, D_INNER, w_out16 + (size_t)ld * D_MODEL * D_INNER, D_INNER,
                hnew, D_MODEL, D_MODEL, D_INNER, d, nullptr);
        }
        add2_kernel<<<(MROWS * D_MODEL) / 256, 256, 0, stream>>>(
            r1, r2, resb, MROWS * D_MODEL);
        hcur = hnew;
    }
    // final residual + LN -> d_out (fp32)
    addln_kernel<<<MROWS, 256, 0, stream>>>(
        hcur, resb, 0, norm_f_w, norm_f_b, nullptr, nullptr, (float*)d_out);
}
